// SlotAttention_17575006175529
// MI455X (gfx1250) — matrix operand order repaired and checked by reading, not compiled, not run on hardware
//
#include <hip/hip_runtime.h>

typedef __attribute__((ext_vector_type(16))) __bf16 v16bf;
typedef __attribute__((ext_vector_type(8)))  __bf16 v8bf;
typedef __attribute__((ext_vector_type(8)))  float  v8f;

#define TS   64
#define TD   4096
#define NB   8
#define NH   16
#define CC   64
#define HC   1024
#define CHUNK 256

static __device__ inline v8f wmma_bf16(v16bf a, v16bf b, v8f c) {
  return __builtin_amdgcn_wmma_f32_16x16x32_bf16(false, a, false, b, (short)0, c, false, false);
}

// A-fragment: two adjacent 16B chunks (K and K+8) merged into one v16bf
static __device__ inline v16bf make_frag(const __bf16* p0, const __bf16* p1) {
  v8bf lo = *(const v8bf*)p0;
  v8bf hi = *(const v8bf*)p1;
  return __builtin_shufflevector(lo, hi, 0,1,2,3,4,5,6,7,8,9,10,11,12,13,14,15);
}

// ---------------- fp32 -> bf16 conversion (vectorized) ----------------
__global__ void cvt_f32_bf16(const float* __restrict__ in, __bf16* __restrict__ out, int n) {
  int i = (blockIdx.x * blockDim.x + threadIdx.x) * 4;
  if (i < n) {
    float4 f = *(const float4*)(in + i);
    out[i + 0] = (__bf16)f.x;
    out[i + 1] = (__bf16)f.y;
    out[i + 2] = (__bf16)f.z;
    out[i + 3] = (__bf16)f.w;
  }
}

// ---------------- GEMM: O = X(M,1024) @ W(1024,1024)^T + bias ----------------
// store_mode 0: O row-major [m][1024] bf16 (q, k)
// store_mode 1: O = V^T per (b,h): O[((bb*16+h)*64+c)*4096 + dd], m = dd*8+bb, n = h*64+c
__global__ __launch_bounds__(256) void gemm_qkv(const __bf16* __restrict__ X,
                                                const __bf16* __restrict__ W,
                                                const float*  __restrict__ bias,
                                                __bf16* __restrict__ O,
                                                int M, int store_mode) {
  int tid  = threadIdx.x;
  int wave = tid >> 5;
  int lane = tid & 31;
  int lm   = lane & 15;
  int lh   = lane >> 4;

  int m0 = blockIdx.x * 128 + wave * 16;   // 8 waves stacked in M
  int n0 = blockIdx.y * 64;                // 4 N-tiles per wave

  const __bf16* xrow = X + (size_t)(m0 + lm) * HC;

  v8f acc[4] = {};
  for (int k0 = 0; k0 < HC; k0 += 32) {
    const __bf16* ap = xrow + k0 + (lh << 4);
    v16bf af = make_frag(ap, ap + 8);
#pragma unroll
    for (int j = 0; j < 4; ++j) {
      const __bf16* bp = W + (size_t)(n0 + j * 16 + lm) * HC + k0 + (lh << 4);
      v16bf bf = *(const v16bf*)bp;
      acc[j] = wmma_bf16(af, bf, acc[j]);
    }
  }

#pragma unroll
  for (int j = 0; j < 4; ++j) {
    int n = n0 + j * 16 + lm;
    float bv = bias[n];
#pragma unroll
    for (int r = 0; r < 8; ++r) {
      int m = m0 + r + (lh << 3);
      float val = acc[j][r] + bv;
      if (store_mode == 0) {
        O[(size_t)m * HC + n] = (__bf16)val;
      } else {
        int dd = m >> 3, bb = m & 7;
        int hh = n >> 6, cc = n & 63;
        O[((size_t)((bb * NH + hh) * CC + cc)) * TD + dd] = (__bf16)val;
      }
    }
  }
}

// ---------------- fused slot-attention core, one block per (b,h) ----------------
__global__ __launch_bounds__(256) void attn_kernel(const __bf16* __restrict__ qb,  // (512,1024)
                                                   const __bf16* __restrict__ kb,  // (32768,1024)
                                                   const __bf16* __restrict__ vt,  // [b][h][c][d]
                                                   float* __restrict__ out) {      // (64,8,1024)
  __shared__ __align__(16) __bf16 a_stage[TS][CHUNK];  // 32 KB
  __shared__ float n_lds[TS];

  int bh = blockIdx.x;
  int bb = bh >> 4;
  int h  = bh & 15;
  int tid  = threadIdx.x;
  int wave = tid >> 5;
  int lane = tid & 31;
  int lm   = lane & 15;
  int lh   = lane >> 4;

  if (tid < TS) n_lds[tid] = 0.001f;
  __syncthreads();

  // Preload Q A-fragments: 4 M-tiles x 2 K-steps (c = 64)
  v16bf qa[4][2];
#pragma unroll
  for (int t = 0; t < 4; ++t) {
    int m = t * 16 + lm;
    const __bf16* qrow = qb + (size_t)(m * NB + bb) * HC + h * CC;
#pragma unroll
    for (int ks = 0; ks < 2; ++ks) {
      int koff = ks * 32 + (lh << 4);
      qa[t][ks] = make_frag(qrow + koff, qrow + koff + 8);
    }
  }

  v8f num[2];
  num[0] = (v8f){};
  num[1] = (v8f){};
  float np[4][8] = {};

  for (int d0 = 0; d0 < TD; d0 += CHUNK) {
    // ---- phase 1: scores + column softmax (over all 64 slots) ----
#pragma unroll
    for (int nt = 0; nt < 2; ++nt) {
      int dloc_base = (wave * 2 + nt) * 16;
      int dglob = d0 + dloc_base + lm;
      const __bf16* krow = kb + (size_t)(dglob * NB + bb) * HC + h * CC;
      v16bf kb0 = *(const v16bf*)(krow + (lh << 4));
      v16bf kb1 = *(const v16bf*)(krow + 32 + (lh << 4));
      v8f sacc[4];
#pragma unroll
      for (int t = 0; t < 4; ++t) {
        v8f cc0 = {};
        cc0 = wmma_bf16(qa[t][0], kb0, cc0);
        cc0 = wmma_bf16(qa[t][1], kb1, cc0);
        sacc[t] = cc0;
      }
      float vmax = -1e30f;
#pragma unroll
      for (int t = 0; t < 4; ++t)
#pragma unroll
        for (int r = 0; r < 8; ++r) {
          sacc[t][r] *= 0.125f;                 // 1/sqrt(64)
          vmax = fmaxf(vmax, sacc[t][r]);
        }
      vmax = fmaxf(vmax, __shfl_xor(vmax, 16, 32));  // other half holds M=+8 of same column
      float dsum = 0.0f;
#pragma unroll
      for (int t = 0; t < 4; ++t)
#pragma unroll
        for (int r = 0; r < 8; ++r) {
          float e = __expf(sacc[t][r] - vmax);
          sacc[t][r] = e;
          dsum += e;
        }
      dsum += __shfl_xor(dsum, 16, 32);
      float rden = 1.0f / dsum;
#pragma unroll
      for (int t = 0; t < 4; ++t)
#pragma unroll
        for (int r = 0; r < 8; ++r) {
          float a = sacc[t][r] * rden;
          np[t][r] += a;                        // partial row-sum (n)
          a_stage[t * 16 + r + (lh << 3)][dloc_base + lm] = (__bf16)a;
        }
    }
    __syncthreads();

    // ---- phase 2: num += a_chunk (64 x 256) @ V_chunk (256 x 64) ----
#pragma unroll
    for (int i = 0; i < 2; ++i) {
      int idx = wave + i * 8;
      int tm = idx >> 2, tn = idx & 3;
      const __bf16* vrow = vt + ((size_t)((bb * NH + h) * CC + tn * 16 + lm)) * TD + d0;
      v8f acc = num[i];
#pragma unroll
      for (int ks = 0; ks < CHUNK / 32; ++ks) {
        const __bf16* ap = &a_stage[tm * 16 + lm][ks * 32 + (lh << 4)];
        v16bf af = make_frag(ap, ap + 8);
        v16bf bf = *(const v16bf*)(vrow + ks * 32 + (lh << 4));
        acc = wmma_bf16(af, bf, acc);
      }
      num[i] = acc;
    }
    __syncthreads();  // before next chunk overwrites a_stage
  }

  // ---- reduce n across lanes & waves ----
#pragma unroll
  for (int t = 0; t < 4; ++t)
#pragma unroll
    for (int r = 0; r < 8; ++r) {
      float v = np[t][r];
      v += __shfl_xor(v, 1, 32);
      v += __shfl_xor(v, 2, 32);
      v += __shfl_xor(v, 4, 32);
      v += __shfl_xor(v, 8, 32);
      if (lm == 0) atomicAdd(&n_lds[t * 16 + r + (lh << 3)], v);
    }
  __syncthreads();

  // ---- normalize + write out (fp32) ----
#pragma unroll
  for (int i = 0; i < 2; ++i) {
    int idx = wave + i * 8;
    int tm = idx >> 2, tn = idx & 3;
#pragma unroll
    for (int r = 0; r < 8; ++r) {
      int srow = tm * 16 + r + (lh << 3);
      int ccol = tn * 16 + lm;
      float val = num[i][r] / n_lds[srow];
      out[((size_t)(srow * NB + bb)) * HC + h * CC + ccol] = val;
    }
  }
}

extern "C" void kernel_launch(void* const* d_in, const int* in_sizes, int n_in,
                              void* d_out, int out_size, void* d_ws, size_t ws_size,
                              hipStream_t stream) {
  const float* s  = (const float*)d_in[0];   // (64, 8, 1024)
  const float* d  = (const float*)d_in[1];   // (4096, 8, 1024)
  const float* Wq = (const float*)d_in[2];
  const float* bq = (const float*)d_in[3];
  const float* Wk = (const float*)d_in[4];
  const float* bk = (const float*)d_in[5];
  const float* Wv = (const float*)d_in[6];
  const float* bv = (const float*)d_in[7];
  float* out = (float*)d_out;

  char* ws = (char*)d_ws;
  size_t off = 0;
  auto alloc = [&](size_t bytes) { char* p = ws + off; off += (bytes + 255) & ~(size_t)255; return p; };

  __bf16* d_bf  = (__bf16*)alloc((size_t)TD * NB * HC * 2);   // 64 MB
  __bf16* s_bf  = (__bf16*)alloc((size_t)TS * NB * HC * 2);   // 1 MB
  __bf16* wq_bf = (__bf16*)alloc((size_t)HC * HC * 2);        // 2 MB
  __bf16* wk_bf = (__bf16*)alloc((size_t)HC * HC * 2);
  __bf16* wv_bf = (__bf16*)alloc((size_t)HC * HC * 2);
  __bf16* q_bf  = (__bf16*)alloc((size_t)TS * NB * HC * 2);   // 1 MB
  __bf16* k_bf  = (__bf16*)alloc((size_t)TD * NB * HC * 2);   // 64 MB
  __bf16* v_t   = (__bf16*)alloc((size_t)TD * NB * HC * 2);   // 64 MB, [b][h][c][d]

  // fp32 -> bf16
  {
    int n;
    n = TD * NB * HC; cvt_f32_bf16<<<n / 1024, 256, 0, stream>>>(d, d_bf, n);
    n = TS * NB * HC; cvt_f32_bf16<<<n / 1024, 256, 0, stream>>>(s, s_bf, n);
    n = HC * HC;      cvt_f32_bf16<<<n / 1024, 256, 0, stream>>>(Wq, wq_bf, n);
    n = HC * HC;      cvt_f32_bf16<<<n / 1024, 256, 0, stream>>>(Wk, wk_bf, n);
    n = HC * HC;      cvt_f32_bf16<<<n / 1024, 256, 0, stream>>>(Wv, wv_bf, n);
  }

  // projections
  gemm_qkv<<<dim3(TS * NB / 128, HC / 64), 256, 0, stream>>>(s_bf, wq_bf, bq, q_bf, TS * NB, 0);
  gemm_qkv<<<dim3(TD * NB / 128, HC / 64), 256, 0, stream>>>(d_bf, wk_bf, bk, k_bf, TD * NB, 0);
  gemm_qkv<<<dim3(TD * NB / 128, HC / 64), 256, 0, stream>>>(d_bf, wv_bf, bv, v_t,  TD * NB, 1);

  // fused attention: one block per (b, h)
  attn_kernel<<<NB * NH, 256, 0, stream>>>(q_bf, k_bf, v_t, out);
}
